// MSAAttention_4286377361473
// MI455X (gfx1250) — compile-verified, matrix operand order checked by reading, not run on hardware
//
#include <hip/hip_runtime.h>
#include <hip/hip_bf16.h>

typedef _Float16 h16;
typedef __attribute__((ext_vector_type(16))) _Float16 v16h;
typedef __attribute__((ext_vector_type(8)))  _Float16 v8h;
typedef __attribute__((ext_vector_type(8)))  float    v8f;

#define S_DIM 128
#define N_DIM 256
#define CM    256
#define CZ    128
#define HEADS 8
#define CH    32
#define ROWS  (S_DIM * N_DIM)   // 32768 MSA rows after flattening (s, residue)

#if defined(__has_builtin)
#if __has_builtin(__builtin_amdgcn_tensor_load_to_lds) && \
    __has_builtin(__builtin_amdgcn_s_wait_tensorcnt)
#define USE_TDM 1
#endif
#endif
#ifndef USE_TDM
#define USE_TDM 0
#endif

// ---------- wave reduction helpers (wave32) ----------
__device__ __forceinline__ float wave32_sum(float v) {
#pragma unroll
  for (int m = 16; m >= 1; m >>= 1) v += __shfl_xor(v, m, 32);
  return v;
}
// reductions within each 16-lane half (rows differ between halves in C layout)
__device__ __forceinline__ float half16_max(float v) {
#pragma unroll
  for (int m = 8; m >= 1; m >>= 1) v = fmaxf(v, __shfl_xor(v, m, 32));
  return v;
}
__device__ __forceinline__ float half16_sum(float v) {
#pragma unroll
  for (int m = 8; m >= 1; m >>= 1) v += __shfl_xor(v, m, 32);
  return v;
}

// concatenate two contiguous 8-half runs into a WMMA fragment register pair
__device__ __forceinline__ v16h cat8(v8h lo, v8h hi) {
  return __builtin_shufflevector(lo, hi, 0, 1, 2, 3, 4, 5, 6, 7,
                                 8, 9, 10, 11, 12, 13, 14, 15);
}
// fragment element e maps to K = (e&7) + (e>=8 ? 16 : 0) + sel*8, i.e. two
// 16-byte contiguous runs at K offsets sel*8 and 16+sel*8 when K is the
// contiguous axis -> load each fragment as two b128 loads.
__device__ __forceinline__ v16h frag_ld(const h16* p /*&row[0]*/, int sel) {
  v8h lo = *reinterpret_cast<const v8h*>(p + sel * 8);
  v8h hi = *reinterpret_cast<const v8h*>(p + 16 + sel * 8);
  return cat8(lo, hi);
}

#if USE_TDM
typedef __attribute__((ext_vector_type(4))) unsigned int u32x4;
typedef __attribute__((ext_vector_type(8))) int          i32x8;
typedef __attribute__((ext_vector_type(4))) int          i32x4;

// Tensor Data Mover: 2D tile of 16-bit elements, global -> LDS.
// D# bitfields per CDNA5 ISA:
//   g0: [1:0]=count=1, [63:32]=lds_addr, [120:64]=global_addr, [127:126]=type=2
//   g1: [17:16]=data_size(1=2B), [79:48]=tensor_dim0, [111:80]=tensor_dim1,
//       [127:112]=tile_dim0, [143:128]=tile_dim1, [207:160]=tensor_dim0_stride
__device__ __forceinline__ void tdm_load_2d(unsigned lds_off, const void* gsrc,
                                            unsigned tile_w, unsigned tile_h,
                                            unsigned row_stride_elems) {
  unsigned long long ga = (unsigned long long)(uintptr_t)gsrc;
  u32x4 g0;
  g0[0] = 1u;                                            // count=1, user-mode load
  g0[1] = lds_off;                                       // lds_addr (bytes)
  g0[2] = (unsigned)ga;                                  // global_addr[31:0]
  g0[3] = (unsigned)((ga >> 32) & 0x01FFFFFFull) | (2u << 30);  // addr hi | type=2
  i32x8 g1;
  g1[0] = (int)(1u << 16);                               // workgroup_mask=0, data_size=2B
  g1[1] = (int)((tile_w & 0xFFFFu) << 16);               // tensor_dim0 lo16
  g1[2] = (int)(((tile_w >> 16) & 0xFFFFu) |             // tensor_dim0 hi16
                ((tile_h & 0xFFFFu) << 16));             // tensor_dim1 lo16
  g1[3] = (int)(((tile_h >> 16) & 0xFFFFu) |             // tensor_dim1 hi16
                ((tile_w & 0xFFFFu) << 16));             // tile_dim0
  g1[4] = (int)(tile_h & 0xFFFFu);                       // tile_dim1; tile_dim2=0
  g1[5] = (int)row_stride_elems;                         // tensor_dim0_stride[31:0]
  g1[6] = 0;                                             // stride0 hi / stride1 lo
  g1[7] = 0;
  i32x4 gz4 = {0, 0, 0, 0};
  i32x8 gz8 = {0, 0, 0, 0, 0, 0, 0, 0};
  // 6-arg form (clang-23 / therock HIP headers)
  __builtin_amdgcn_tensor_load_to_lds(g0, g1, gz4, gz4, gz8, 0);
}
#endif

// Stage a (tile_h x tile_w)-element f16 tile (dst rows contiguous) from global
// (row stride row_stride elems) into LDS.  TDM path: single tensor_load_to_lds
// issued by wave 0; fallback: vector copies.  Caller must __syncthreads after.
__device__ __forceinline__ void stage_tile(h16* lds_dst, const h16* gsrc,
                                           int tile_w, int tile_h,
                                           int row_stride, int tid) {
#if USE_TDM
  if (tid < 32) {
    tdm_load_2d((unsigned)(uintptr_t)lds_dst, gsrc,
                (unsigned)tile_w, (unsigned)tile_h, (unsigned)row_stride);
    __builtin_amdgcn_s_wait_tensorcnt(0);
  }
#else
  int per = tile_w >> 3;
  for (int t = tid; t < tile_h * per; t += 256) {
    int r = t / per, c8 = (t - r * per) * 8;
    *reinterpret_cast<uint4*>(lds_dst + (size_t)r * tile_w + c8) =
        *reinterpret_cast<const uint4*>(gsrc + (size_t)r * row_stride + c8);
  }
#endif
}

// ---------------------------------------------------------------------------
// Kernel 0: one-shot weight pre-transpose: five 256x256 f32 matrices ->
// f16 W^T[n][k] (K contiguous), so every hot-loop B-fragment stage is a pure
// block copy.  64x64 tiles bounced through LDS.
// ---------------------------------------------------------------------------
__global__ void wtrans_kernel(const float* __restrict__ wq,
                              const float* __restrict__ wk,
                              const float* __restrict__ wv,
                              const float* __restrict__ wg,
                              const float* __restrict__ wo,
                              h16* __restrict__ Wt) {
  __shared__ alignas(16) h16 tl[64][64];
  int wsel = blockIdx.y;
  const float* W = (wsel == 0) ? wq : (wsel == 1) ? wk
                 : (wsel == 2) ? wv : (wsel == 3) ? wg : wo;
  h16* Wo = Wt + (size_t)wsel * CM * CM;
  int n0 = (blockIdx.x & 3) * 64, k0 = (blockIdx.x >> 2) * 64;
  int tid = threadIdx.x;
  for (int t = tid; t < 64 * 64 / 4; t += 256) {
    int kk = t >> 4, n4 = (t & 15) * 4;
    float4 w4 = *reinterpret_cast<const float4*>(W + (size_t)(k0 + kk) * CM + n0 + n4);
    tl[n4 + 0][kk] = (h16)w4.x;
    tl[n4 + 1][kk] = (h16)w4.y;
    tl[n4 + 2][kk] = (h16)w4.z;
    tl[n4 + 3][kk] = (h16)w4.w;
  }
  __syncthreads();
  for (int t = tid; t < 64 * 64 / 8; t += 256) {
    int n = t >> 3, k8 = (t & 7) * 8;
    *reinterpret_cast<uint4*>(Wo + (size_t)(n0 + n) * CM + k0 + k8) =
        *reinterpret_cast<uint4*>(&tl[n][k8]);
  }
}

// ---------------------------------------------------------------------------
// Kernel 1: pair bias  zb[h][i][j] = LN(z[i,j,:]) @ w_z[:,h]
// one wave per (i,j) pair; 128 channels -> 4 contiguous per lane (float4)
// ---------------------------------------------------------------------------
__global__ void pair_bias_kernel(const float* __restrict__ z,
                                 const float* __restrict__ lnw,
                                 const float* __restrict__ lnb,
                                 const float* __restrict__ wz,
                                 float* __restrict__ zb) {
  int wv = threadIdx.x >> 5, lane = threadIdx.x & 31;
  int p = blockIdx.x * 8 + wv;  // pair index 0..65535
  const float* zr = z + (size_t)p * CZ;
  float4 x4 = *reinterpret_cast<const float4*>(zr + lane * 4);
  float x[4] = {x4.x, x4.y, x4.z, x4.w};
  float s = 0.f, s2 = 0.f;
#pragma unroll
  for (int t = 0; t < 4; ++t) { s += x[t]; s2 += x[t] * x[t]; }
  s = wave32_sum(s); s2 = wave32_sum(s2);
  float mu = s * (1.f / CZ);
  float var = s2 * (1.f / CZ) - mu * mu;
  float rstd = rsqrtf(var + 1e-5f);
  float4 w4 = *reinterpret_cast<const float4*>(lnw + lane * 4);
  float4 b4 = *reinterpret_cast<const float4*>(lnb + lane * 4);
  float lw[4] = {w4.x, w4.y, w4.z, w4.w};
  float lb[4] = {b4.x, b4.y, b4.z, b4.w};
  float ph[HEADS] = {0, 0, 0, 0, 0, 0, 0, 0};
#pragma unroll
  for (int t = 0; t < 4; ++t) {
    int c = lane * 4 + t;
    float xn = (x[t] - mu) * rstd * lw[t] + lb[t];
    const float4* wzr = reinterpret_cast<const float4*>(wz + c * HEADS);
    float4 wa = wzr[0], wb = wzr[1];
    ph[0] += xn * wa.x; ph[1] += xn * wa.y; ph[2] += xn * wa.z; ph[3] += xn * wa.w;
    ph[4] += xn * wb.x; ph[5] += xn * wb.y; ph[6] += xn * wb.z; ph[7] += xn * wb.w;
  }
#pragma unroll
  for (int hh = 0; hh < HEADS; ++hh) ph[hh] = wave32_sum(ph[hh]);
  if (lane == 0) {
#pragma unroll
    for (int hh = 0; hh < HEADS; ++hh)
      zb[(size_t)hh * N_DIM * N_DIM + p] = ph[hh];
  }
}

// ---------------------------------------------------------------------------
// Kernel 2: LayerNorm(m) + one of {Q,K,V,G} projection (blockIdx.y selects).
// Weight chunks staged from pre-transposed Wt (TDM tensor_load_to_lds or
// vector copies).  V projection stored transposed per head.
// ---------------------------------------------------------------------------
__global__ void ln_qkvg_kernel(const float* __restrict__ m,
                               const float* __restrict__ lnw,
                               const float* __restrict__ lnb,
                               const h16* __restrict__ Wt,
                               const float* __restrict__ bg,
                               h16* __restrict__ Qh, h16* __restrict__ Kh,
                               h16* __restrict__ Vt, h16* __restrict__ Gh) {
  __shared__ alignas(16) h16 ml[64][CM];    // 32 KB LN'ed rows, K contiguous
  __shared__ alignas(16) h16 wlt[CM][32];   // 16 KB weight chunk (transposed)
  int tid = threadIdx.x, wvid = tid >> 5, lane = tid & 31;
  int proj = blockIdx.y;
  const h16* Wtp = Wt + (size_t)proj * CM * CM;
  int row0 = blockIdx.x * 64;

  // --- LayerNorm: 8 rows per wave, 8 contiguous channels per lane ---
  for (int q = 0; q < 8; ++q) {
    int r = wvid * 8 + q;
    const float* mr = m + (size_t)(row0 + r) * CM + lane * 8;
    float4 xa = *reinterpret_cast<const float4*>(mr);
    float4 xb = *reinterpret_cast<const float4*>(mr + 4);
    float x[8] = {xa.x, xa.y, xa.z, xa.w, xb.x, xb.y, xb.z, xb.w};
    float s = 0.f, s2 = 0.f;
#pragma unroll
    for (int t = 0; t < 8; ++t) { s += x[t]; s2 += x[t] * x[t]; }
    s = wave32_sum(s); s2 = wave32_sum(s2);
    float mu = s * (1.f / CM);
    float var = s2 * (1.f / CM) - mu * mu;
    float rstd = rsqrtf(var + 1e-5f);
    float4 wa = *reinterpret_cast<const float4*>(lnw + lane * 8);
    float4 wb = *reinterpret_cast<const float4*>(lnw + lane * 8 + 4);
    float4 ba = *reinterpret_cast<const float4*>(lnb + lane * 8);
    float4 bb = *reinterpret_cast<const float4*>(lnb + lane * 8 + 4);
    float lw[8] = {wa.x, wa.y, wa.z, wa.w, wb.x, wb.y, wb.z, wb.w};
    float lb[8] = {ba.x, ba.y, ba.z, ba.w, bb.x, bb.y, bb.z, bb.w};
    v8h xn;
#pragma unroll
    for (int t = 0; t < 8; ++t)
      xn[t] = (h16)((x[t] - mu) * rstd * lw[t] + lb[t]);
    *reinterpret_cast<v8h*>(&ml[r][lane * 8]) = xn;   // one ds_store_b128
  }

  // --- GEMM: wave -> (m-tile, 8 n-tiles) ---
  int mt = wvid & 3;       // 4 M-tiles of 16 rows
  int ng = wvid >> 2;      // 2 groups of 8 N-tiles
  int ln = lane & 15, sel = lane >> 4;
  v8f acc[8] = {};
  for (int k0 = 0; k0 < CM; k0 += 32) {
    __syncthreads();
    // Wt[n][k0..k0+31] -> wlt[n][0..31] : 256 rows x 32 halfs, stride 256
    stage_tile(&wlt[0][0], Wtp + k0, 32, CM, CM, tid);
    __syncthreads();
    v16h a = frag_ld(&ml[mt * 16 + ln][k0], sel);
    v16h b[8];
#pragma unroll
    for (int i = 0; i < 8; ++i)
      b[i] = frag_ld(&wlt[(ng * 8 + i) * 16 + ln][0], sel);
#pragma unroll
    for (int i = 0; i < 8; ++i)
      acc[i] = __builtin_amdgcn_wmma_f32_16x16x32_f16(false, a, false, b[i],
                                                      (short)0, acc[i], false, false);
  }

  const float scaleq = 0.17677669529663688f;  // CH^-0.5
  h16* O = (proj == 0) ? Qh : (proj == 1) ? Kh : (proj == 2) ? Vt : Gh;
#pragma unroll
  for (int i = 0; i < 8; ++i) {
    int nt = ng * 8 + i;
    int col = nt * 16 + ln;
#pragma unroll
    for (int vi = 0; vi < 8; ++vi) {
      int grow = row0 + mt * 16 + vi + 8 * sel;
      float v = acc[i][vi];
      if (proj == 0) v *= scaleq;          // fold q scaling
      if (proj == 3) v += bg[col];         // fold gating bias
      if (proj == 2) {
        // transposed per (s, head): Vt[(s*256 + col)*256 + residue]
        O[((size_t)(grow >> 8) * CM + col) * N_DIM + (grow & 255)] = (h16)v;
      } else {
        O[(size_t)grow * CM + col] = (h16)v;
      }
    }
  }
}

// ---------------------------------------------------------------------------
// Kernel 3: fused attention per (s, head).  q/k tiles row-major (K=c
// contiguous); v staged from pre-transposed Vt.  Tiles staged with the TDM
// (tensor_load_to_lds) when available.  Flash-style online softmax.
// ---------------------------------------------------------------------------
__global__ void attn_kernel(const h16* __restrict__ Qh, const h16* __restrict__ Kh,
                            const h16* __restrict__ Vt, const h16* __restrict__ Gh,
                            const float* __restrict__ zb,
                            const float* __restrict__ mask,
                            h16* __restrict__ OH) {
  __shared__ alignas(16) h16 qs[N_DIM][CH];    // 16 KB [row][c]
  __shared__ alignas(16) h16 ks[N_DIM][CH];    // 16 KB [row][c]
  __shared__ alignas(16) h16 vst[CH][N_DIM];   // 16 KB transposed [ch][j]
  __shared__ alignas(16) h16 pst[8][16][CH];   // per-wave p relayout, 8 KB
  __shared__ float maskb[N_DIM];               // 1 KB
  int s = blockIdx.x, hh = blockIdx.y;
  int tid = threadIdx.x, wv = tid >> 5, lane = tid & 31;
  int ln = lane & 15, sel = lane >> 4;
  const float* zbh = zb + (size_t)hh * N_DIM * N_DIM;
  size_t base = (size_t)s * N_DIM * (HEADS * CH) + (size_t)hh * CH;

  // q/k: 256 rows x 32 halfs, global row stride 256; v: contiguous 32x256
  stage_tile(&qs[0][0], Qh + base, CH, N_DIM, HEADS * CH, tid);
  stage_tile(&ks[0][0], Kh + base, CH, N_DIM, HEADS * CH, tid);
  stage_tile(&vst[0][0], Vt + ((size_t)s * CM + hh * CH) * N_DIM,
             N_DIM, CH, N_DIM, tid);
  maskb[tid] = 1e9f * (mask[s * N_DIM + tid] - 1.0f);
  __syncthreads();

  for (int qt = wv; qt < 16; qt += 8) {   // 2 query tiles (16 rows) per wave
    v16h aq = frag_ld(&qs[qt * 16 + ln][0], sel);
    // warm GL2/WGP$ for the pair-bias rows this tile will consume
    __builtin_prefetch(zbh + (qt * 16 + ln) * N_DIM, 0, 1);
    float mrow[8], lrow[8];
#pragma unroll
    for (int vi = 0; vi < 8; ++vi) { mrow[vi] = -1e30f; lrow[vi] = 0.f; }
    v8f o0 = {}, o1 = {};

    for (int kb = 0; kb < 8; ++kb) {      // 32 keys per block
      int j0 = kb * 32;
      // kᵀ B fragments: K dim = c is contiguous in ks rows
      v16h bk0 = frag_ld(&ks[j0 + ln][0], sel);
      v16h bk1 = frag_ld(&ks[j0 + 16 + ln][0], sel);
      v8f zc = {};
      v8f s0 = __builtin_amdgcn_wmma_f32_16x16x32_f16(false, aq, false, bk0,
                                                      (short)0, zc, false, false);
      v8f s1 = __builtin_amdgcn_wmma_f32_16x16x32_f16(false, aq, false, bk1,
                                                      (short)0, zc, false, false);
#pragma unroll
      for (int vi = 0; vi < 8; ++vi) {
        int qrow = qt * 16 + vi + 8 * sel;
        int c0 = j0 + ln, c1 = c0 + 16;
        float v0 = s0[vi] + zbh[qrow * N_DIM + c0] + maskb[c0];
        float v1 = s1[vi] + zbh[qrow * N_DIM + c1] + maskb[c1];
        float tmax = half16_max(fmaxf(v0, v1));
        float mn_ = fmaxf(mrow[vi], tmax);
        float p0 = __expf(v0 - mn_), p1 = __expf(v1 - mn_);
        float rs = half16_sum(p0 + p1);
        float scl = __expf(mrow[vi] - mn_);
        mrow[vi] = mn_;
        lrow[vi] = lrow[vi] * scl + rs;
        o0[vi] *= scl; o1[vi] *= scl;
        pst[wv][vi + 8 * sel][ln]      = (h16)p0;   // C-layout -> LDS
        pst[wv][vi + 8 * sel][ln + 16] = (h16)p1;
      }
      // same-wave LDS RAW: DS ops from one wave stay in order
      v16h ap  = frag_ld(&pst[wv][ln][0], sel);
      // P·V B fragments: K dim = residue is contiguous in vst rows
      v16h bv0 = frag_ld(&vst[ln][j0], sel);
      v16h bv1 = frag_ld(&vst[ln + 16][j0], sel);
      o0 = __builtin_amdgcn_wmma_f32_16x16x32_f16(false, ap, false, bv0,
                                                  (short)0, o0, false, false);
      o1 = __builtin_amdgcn_wmma_f32_16x16x32_f16(false, ap, false, bv1,
                                                  (short)0, o1, false, false);
    }

#pragma unroll
    for (int vi = 0; vi < 8; ++vi) {
      int qrow = qt * 16 + vi + 8 * sel;
      float inv = 1.f / lrow[vi];
      size_t rb = base + (size_t)qrow * (HEADS * CH);
      float g0 = (float)Gh[rb + ln];
      float g1 = (float)Gh[rb + ln + 16];
      g0 = 1.f / (1.f + __expf(-g0));
      g1 = 1.f / (1.f + __expf(-g1));
      OH[rb + ln]      = (h16)(o0[vi] * inv * g0);
      OH[rb + ln + 16] = (h16)(o1[vi] * inv * g1);
    }
  }
}

// ---------------------------------------------------------------------------
// Kernel 4: output projection  out = OH(f16) @ w_o + b_o   (f32 out)
// ---------------------------------------------------------------------------
__global__ void out_proj_kernel(const h16* __restrict__ OH,
                                const h16* __restrict__ Wto,   // pre-transposed
                                const float* __restrict__ bo,
                                float* __restrict__ out) {
  __shared__ alignas(16) h16 al[64][CM];    // 32 KB activations, K contiguous
  __shared__ alignas(16) h16 wlt[CM][32];   // 16 KB weight chunk (transposed)
  int tid = threadIdx.x, wvid = tid >> 5, lane = tid & 31;
  int row0 = blockIdx.x * 64;
  // 64 rows x 256 halfs, contiguous
  stage_tile(&al[0][0], OH + (size_t)row0 * CM, CM, 64, CM, tid);
  int mt = wvid & 3, ng = wvid >> 2;
  int ln = lane & 15, sel = lane >> 4;
  v8f acc[8] = {};
  for (int k0 = 0; k0 < CM; k0 += 32) {
    __syncthreads();
    stage_tile(&wlt[0][0], Wto + k0, 32, CM, CM, tid);
    __syncthreads();
    v16h a = frag_ld(&al[mt * 16 + ln][k0], sel);
    v16h b[8];
#pragma unroll
    for (int i = 0; i < 8; ++i)
      b[i] = frag_ld(&wlt[(ng * 8 + i) * 16 + ln][0], sel);
#pragma unroll
    for (int i = 0; i < 8; ++i)
      acc[i] = __builtin_amdgcn_wmma_f32_16x16x32_f16(false, a, false, b[i],
                                                      (short)0, acc[i], false, false);
  }
#pragma unroll
  for (int i = 0; i < 8; ++i) {
    int nt = ng * 8 + i;
    int col = nt * 16 + ln;
#pragma unroll
    for (int vi = 0; vi < 8; ++vi) {
      int grow = row0 + mt * 16 + vi + 8 * sel;
      out[(size_t)grow * CM + col] = acc[i][vi] + bo[col];
    }
  }
}

// ---------------------------------------------------------------------------
extern "C" void kernel_launch(void* const* d_in, const int* in_sizes, int n_in,
                              void* d_out, int out_size, void* d_ws, size_t ws_size,
                              hipStream_t stream) {
  (void)in_sizes; (void)n_in; (void)out_size; (void)ws_size;
  const float* m      = (const float*)d_in[0];
  const float* z      = (const float*)d_in[1];
  const float* mask   = (const float*)d_in[2];
  const float* ln_m_w = (const float*)d_in[3];
  const float* ln_m_b = (const float*)d_in[4];
  const float* ln_z_w = (const float*)d_in[5];
  const float* ln_z_b = (const float*)d_in[6];
  const float* w_z    = (const float*)d_in[7];
  const float* w_q    = (const float*)d_in[8];
  const float* w_k    = (const float*)d_in[9];
  const float* w_v    = (const float*)d_in[10];
  const float* w_g    = (const float*)d_in[11];
  const float* b_g    = (const float*)d_in[12];
  const float* w_o    = (const float*)d_in[13];
  const float* b_o    = (const float*)d_in[14];

  // workspace: zb f32 (2 MB) | Wt f16 5x[256,256] (640 KB) | Q,K,Vt,G,OH f16
  char* ws = (char*)d_ws;
  size_t off = 0;
  float* zb = (float*)(ws + off); off += (size_t)HEADS * N_DIM * N_DIM * sizeof(float);
  h16* Wt  = (h16*)(ws + off); off += (size_t)5 * CM * CM * sizeof(h16);
  h16* Qh  = (h16*)(ws + off); off += (size_t)ROWS * CM * sizeof(h16);
  h16* Kh  = (h16*)(ws + off); off += (size_t)ROWS * CM * sizeof(h16);
  h16* Vt  = (h16*)(ws + off); off += (size_t)ROWS * CM * sizeof(h16);
  h16* Gh  = (h16*)(ws + off); off += (size_t)ROWS * CM * sizeof(h16);
  h16* OHh = (h16*)(ws + off); off += (size_t)ROWS * CM * sizeof(h16);

  wtrans_kernel<<<dim3(16, 5), 256, 0, stream>>>(
      w_q, w_k, w_v, w_g, w_o, Wt);
  pair_bias_kernel<<<dim3(N_DIM * N_DIM / 8), 256, 0, stream>>>(
      z, ln_z_w, ln_z_b, w_z, zb);
  ln_qkvg_kernel<<<dim3(ROWS / 64, 4), 256, 0, stream>>>(
      m, ln_m_w, ln_m_b, Wt, b_g, Qh, Kh, Vt, Gh);
  attn_kernel<<<dim3(S_DIM, HEADS), 256, 0, stream>>>(
      Qh, Kh, Vt, Gh, zb, mask, OHh);
  out_proj_kernel<<<dim3(ROWS / 64), 256, 0, stream>>>(
      OHh, Wt + (size_t)4 * CM * CM, b_o, (float*)d_out);
}
